// LPAENet_61873298866792
// MI455X (gfx1250) — compile-verified, matrix operand order checked by reading, not run on hardware
//
#include <hip/hip_runtime.h>
#include <float.h>

// ---------------------------------------------------------------------------
// MI455X (gfx1250) implementation of the LPAE forward pass.
//
// Roofline: embed GEMM = 8.6 GFLOP reading 134 MB (AI ~63 F/B). At 23.3 TB/s
// the HBM stream of x takes ~5.8us; with v_wmma_f32_16x16x32_f16 (fp32 acc)
// the matrix math falls below that floor, so the embed kernel is a pure HBM
// streamer: double-buffered 64-wide K chunks, fp32->f16 convert in regs,
// fragment-major LDS layout so every WMMA operand load is ds_load_b128.
// All [*,128]@[128,128] GEMMs share a WMMA kernel with fused epilogues.
// N=16 attention and prototype scoring are negligible -> VALU kernels.
// ---------------------------------------------------------------------------

typedef _Float16 f16_t;
typedef _Float16 v16h __attribute__((ext_vector_type(16)));
typedef _Float16 v8h  __attribute__((ext_vector_type(8)));
typedef float    v8f  __attribute__((ext_vector_type(8)));

#define NEG_MAX (-3.402823466e38f)
#define ATT_SCALE 0.17677669529663687f   // 1/sqrt(32)

// --- WMMA helpers (ISA 7.12.2 layouts, wave32) -----------------------------
__device__ __forceinline__ v8f wmma_f16(v16h a, v16h b, v8f c) {
    return __builtin_amdgcn_wmma_f32_16x16x32_f16(false, a, false, b, (short)0, c, false, false);
}

// A (16x32 f16, row-major LDS): element j of lane (m,hb):
//   j<8  -> k = k0 + hb*8 + j        (contiguous 16B run)
//   j>=8 -> k = k0 + 16 + hb*8 + j-8 (contiguous 16B run)
__device__ __forceinline__ v16h frag_a(const f16_t* a, int lda, int k0, int lane) {
    const int m = lane & 15, hb = (lane >> 4) & 1;
    const v8h lo = *(const v8h*)(a + m * lda + k0 + hb * 8);
    const v8h hi = *(const v8h*)(a + m * lda + k0 + 16 + hb * 8);
    return __builtin_shufflevector(lo, hi, 0, 1, 2, 3, 4, 5, 6, 7,
                                           8, 9, 10, 11, 12, 13, 14, 15);
}

// B stored fragment-major: per 32x16 block, lane's 16 f16 are contiguous.
__device__ __forceinline__ v16h frag_b(const f16_t* blockbase, int lane) {
    const v8h lo = *(const v8h*)(blockbase + lane * 16);
    const v8h hi = *(const v8h*)(blockbase + lane * 16 + 8);
    return __builtin_shufflevector(lo, hi, 0, 1, 2, 3, 4, 5, 6, 7,
                                           8, 9, 10, 11, 12, 13, 14, 15);
}

// Scatter index for fragment-major B fill: (k, n) -> linear f16 index.
// Block (kb = k/32, nb = n/16) at (nb*kblocks + kb)*512; inside: lane*16 + j.
__device__ __forceinline__ int bfrag_idx(int k, int n, int kblocks) {
    const int kk = k & 31;
    const int l = (n & 15) + (((kk >> 3) & 1) << 4);
    const int j = (kk & 7) + ((kk >> 4) << 3);
    return (((n >> 4) * kblocks + (k >> 5)) << 9) + (l << 4) + j;
}

// C/D 16x16 f32 scatter to LDS: VGPR r <-> M = r + 8*(lane>=16), N = n0+(lane&15).
__device__ __forceinline__ void store_c(float* cl, v8f c, int n0, int lane) {
    const int n = n0 + (lane & 15);
    const int mofs = (lane >> 4) * 8;
#pragma unroll
    for (int r = 0; r < 8; ++r) cl[(r + mofs) * 128 + n] = c[r];
}

// ---------------------------------------------------------------------------
// Kernel 1: h = relu(LN(x @ w_embed + b)) ; x:[8192,4096] w:[4096,128]
// Double-buffered K stream (64-wide chunks), 8 waves x one 16x16 WMMA tile.
// ---------------------------------------------------------------------------
__global__ __launch_bounds__(256) void embed_gemm_kernel(
    const float* __restrict__ x, const float* __restrict__ w,
    const float* __restrict__ bias, const float* __restrict__ gamma,
    const float* __restrict__ beta, float* __restrict__ out_f32,
    f16_t* __restrict__ out_f16)
{
    __shared__ __align__(16) f16_t Al[2][16 * 64];      // A chunk, row-major
    __shared__ __align__(16) f16_t Bl[2][64 * 128];     // W chunk, fragment-major
    __shared__ float Cl[16 * 128];

    const int tid = threadIdx.x, lane = tid & 31, wv = tid >> 5;
    const int row0 = blockIdx.x * 16;
    const int ar = tid >> 4, ac = (tid & 15) * 4;       // this thread's X float4
    const float* xrow = x + (size_t)(row0 + ar) * 4096 + ac;

    auto stage = [&](int buf, const float4& xv, const float4* wr) {
        f16_t* A = Al[buf];
        f16_t* B = Bl[buf];
        A[ar * 64 + ac + 0] = (f16_t)xv.x;
        A[ar * 64 + ac + 1] = (f16_t)xv.y;
        A[ar * 64 + ac + 2] = (f16_t)xv.z;
        A[ar * 64 + ac + 3] = (f16_t)xv.w;
#pragma unroll
        for (int rr = 0; rr < 8; ++rr) {
            const int idx = (tid + rr * 256) * 4;       // linear in 64x128 chunk
            const int k = idx >> 7, n = idx & 127;
            B[bfrag_idx(k, n + 0, 2)] = (f16_t)wr[rr].x;
            B[bfrag_idx(k, n + 1, 2)] = (f16_t)wr[rr].y;
            B[bfrag_idx(k, n + 2, 2)] = (f16_t)wr[rr].z;
            B[bfrag_idx(k, n + 3, 2)] = (f16_t)wr[rr].w;
        }
    };

    // Prologue: load + stage chunk 0.
    float4 xv = *(const float4*)(xrow);
    float4 wr[8];
#pragma unroll
    for (int rr = 0; rr < 8; ++rr)
        wr[rr] = *(const float4*)(w + (size_t)(tid + rr * 256) * 4);
    stage(0, xv, wr);

    v8f c = {0.f, 0.f, 0.f, 0.f, 0.f, 0.f, 0.f, 0.f};
    int cur = 0;
    for (int kt = 0; kt < 4096; kt += 64) {
        __syncthreads();                                 // stage(cur) visible
        const bool more = (kt + 64) < 4096;
        if (more) {                                      // fetch next chunk -> regs
            xv = *(const float4*)(xrow + kt + 64);
            const float* wp = w + (size_t)(kt + 64) * 128;
#pragma unroll
            for (int rr = 0; rr < 8; ++rr)
                wr[rr] = *(const float4*)(wp + (tid + rr * 256) * 4);
            if (kt + 128 < 4096)                         // global_prefetch_b8
                __builtin_prefetch(xrow + kt + 128, 0, 0);
        }
        const f16_t* ab = Al[cur];
        const f16_t* bb = Bl[cur] + (wv * 2) * 512;      // nb = wv
#pragma unroll
        for (int kb = 0; kb < 2; ++kb)
            c = wmma_f16(frag_a(ab, 64, kb * 32, lane), frag_b(bb + kb * 512, lane), c);
        if (more) stage(cur ^ 1, xv, wr);
        cur ^= 1;
    }
    store_c(Cl, c, wv * 16, lane);
    __syncthreads();

    // Epilogue: bias + LayerNorm + ReLU; one 16-thread group per row.
    const int row = tid >> 4, j = tid & 15;
    float acc[8], s1 = 0.f, s2 = 0.f;
#pragma unroll
    for (int q = 0; q < 8; ++q) {
        const int col = j + q * 16;
        const float a = Cl[row * 128 + col] + bias[col];
        acc[q] = a; s1 += a; s2 += a * a;
    }
#pragma unroll
    for (int m = 8; m >= 1; m >>= 1) { s1 += __shfl_xor(s1, m, 16); s2 += __shfl_xor(s2, m, 16); }
    const float mu = s1 * (1.f / 128.f);
    const float rs = rsqrtf(s2 * (1.f / 128.f) - mu * mu + 1e-5f);
#pragma unroll
    for (int q = 0; q < 8; ++q) {
        const int col = j + q * 16;
        const float y = fmaxf((acc[q] - mu) * rs * gamma[col] + beta[col], 0.f);
        const size_t o = (size_t)(row0 + row) * 128 + col;
        out_f32[o] = y;
        out_f16[o] = (f16_t)y;
    }
}

// ---------------------------------------------------------------------------
// Kernel 2: generic [M,128] @ [128,128] WMMA GEMM with fused epilogue.
//   EPI=0: out = acc + bias
//   EPI=2: out = LN(res + relu(acc + bias))   (safe in-place: block-local rows)
// ---------------------------------------------------------------------------
template <int EPI>
__global__ __launch_bounds__(256) void gemm128_kernel(
    const f16_t* __restrict__ A, const float* __restrict__ W,
    const float* __restrict__ bias, const float* __restrict__ gamma,
    const float* __restrict__ beta, const float* __restrict__ res,
    float* __restrict__ out_f32, f16_t* __restrict__ out_f16)
{
    __shared__ __align__(16) f16_t Al[16 * 128];
    __shared__ __align__(16) f16_t Bl[128 * 128];        // fragment-major
    __shared__ float Cl[16 * 128];

    const int tid = threadIdx.x, lane = tid & 31, wv = tid >> 5;
    const int row0 = blockIdx.x * 16;

    // A tile: 2048 f16 = 256 x v8h (ds_store_b128 path).
    ((v8h*)Al)[tid] = ((const v8h*)(A + (size_t)row0 * 128))[tid];
    // W: 16384 f32 -> f16, scattered into fragment-major order.
#pragma unroll
    for (int rr = 0; rr < 16; ++rr) {
        const int idx = (tid + rr * 256) * 4;
        const int k = idx >> 7, n = idx & 127;
        const float4 w4 = *(const float4*)(W + idx);
        Bl[bfrag_idx(k, n + 0, 4)] = (f16_t)w4.x;
        Bl[bfrag_idx(k, n + 1, 4)] = (f16_t)w4.y;
        Bl[bfrag_idx(k, n + 2, 4)] = (f16_t)w4.z;
        Bl[bfrag_idx(k, n + 3, 4)] = (f16_t)w4.w;
    }
    __syncthreads();

    v8f c = {0.f, 0.f, 0.f, 0.f, 0.f, 0.f, 0.f, 0.f};
    const f16_t* bb = Bl + (wv * 4) * 512;               // nb = wv, 4 k-blocks
#pragma unroll
    for (int kb = 0; kb < 4; ++kb)
        c = wmma_f16(frag_a(Al, 128, kb * 32, lane), frag_b(bb + kb * 512, lane), c);
    store_c(Cl, c, wv * 16, lane);
    __syncthreads();

    const int row = tid >> 4, j = tid & 15;
    if constexpr (EPI == 0) {
#pragma unroll
        for (int q = 0; q < 8; ++q) {
            const int col = j + q * 16;
            const float y = Cl[row * 128 + col] + bias[col];
            const size_t o = (size_t)(row0 + row) * 128 + col;
            out_f32[o] = y;
            if (out_f16) out_f16[o] = (f16_t)y;
        }
    } else {
        float t[8], s1 = 0.f, s2 = 0.f;
#pragma unroll
        for (int q = 0; q < 8; ++q) {
            const int col = j + q * 16;
            const float a = fmaxf(Cl[row * 128 + col] + bias[col], 0.f);
            const float v = res[(size_t)(row0 + row) * 128 + col] + a;
            t[q] = v; s1 += v; s2 += v * v;
        }
#pragma unroll
        for (int m = 8; m >= 1; m >>= 1) { s1 += __shfl_xor(s1, m, 16); s2 += __shfl_xor(s2, m, 16); }
        const float mu = s1 * (1.f / 128.f);
        const float rs = rsqrtf(s2 * (1.f / 128.f) - mu * mu + 1e-5f);
#pragma unroll
        for (int q = 0; q < 8; ++q) {
            const int col = j + q * 16;
            const float y = (t[q] - mu) * rs * gamma[col] + beta[col];
            const size_t o = (size_t)(row0 + row) * 128 + col;
            out_f32[o] = y;
            out_f16[o] = (f16_t)y;
        }
    }
}

// ---------------------------------------------------------------------------
// Kernel 3: SAB attention core. Per batch: masked 4-head softmax(QK^T/s)V,
// h1 = LN(q + o).  N=16 -> pure VALU, negligible cost.
// ---------------------------------------------------------------------------
__global__ __launch_bounds__(256) void sab_attn_kernel(
    const float* __restrict__ q, const float* __restrict__ k, const float* __restrict__ v,
    const int* __restrict__ cate, const float* __restrict__ gamma,
    const float* __restrict__ beta, float* __restrict__ out_f32, f16_t* __restrict__ out_f16)
{
    __shared__ float qs[16 * 128], ks[16 * 128], vs[16 * 128];
    __shared__ float dots[4 * 16 * 16];
    __shared__ float cm[16];
    const int b = blockIdx.x, tid = threadIdx.x;
    const size_t base = (size_t)b * 2048;
#pragma unroll
    for (int rr = 0; rr < 8; ++rr) {
        const int idx = tid + rr * 256;
        qs[idx] = q[base + idx]; ks[idx] = k[base + idx]; vs[idx] = v[base + idx];
    }
    if (tid < 16) cm[tid] = (cate[b * 16 + tid] != -1) ? 1.f : 0.f;
    __syncthreads();
#pragma unroll
    for (int e = 0; e < 4; ++e) {
        const int id = tid * 4 + e;                      // (head, i, j)
        const int hh = id >> 8, i = (id >> 4) & 15, j = id & 15;
        float s = 0.f;
#pragma unroll
        for (int d = 0; d < 32; ++d) s += qs[i * 128 + hh * 32 + d] * ks[j * 128 + hh * 32 + d];
        s *= ATT_SCALE;
        if (cm[i] * cm[j] != 1.f) s = NEG_MAX;
        dots[id] = s;
    }
    __syncthreads();
    if (tid < 64) {                                      // softmax over each (head,i) row
        float* rp = dots + tid * 16;
        float mx = NEG_MAX;
        for (int j = 0; j < 16; ++j) mx = fmaxf(mx, rp[j]);
        float tmp[16], sum = 0.f;
        for (int j = 0; j < 16; ++j) { tmp[j] = __expf(rp[j] - mx); sum += tmp[j]; }
        const float inv = 1.f / sum;
        for (int j = 0; j < 16; ++j) rp[j] = tmp[j] * inv;
    }
    __syncthreads();
    const int row = tid >> 4, jj = tid & 15;
    float t[8], s1 = 0.f, s2 = 0.f;
#pragma unroll
    for (int qq = 0; qq < 8; ++qq) {
        const int col = jj + qq * 16, hh = col >> 5;
        float o = 0.f;
#pragma unroll
        for (int j = 0; j < 16; ++j) o += dots[(hh * 16 + row) * 16 + j] * vs[j * 128 + col];
        const float h1 = qs[row * 128 + col] + o;
        t[qq] = h1; s1 += h1; s2 += h1 * h1;
    }
#pragma unroll
    for (int m = 8; m >= 1; m >>= 1) { s1 += __shfl_xor(s1, m, 16); s2 += __shfl_xor(s2, m, 16); }
    const float mu = s1 * (1.f / 128.f);
    const float rs = rsqrtf(s2 * (1.f / 128.f) - mu * mu + 1e-5f);
#pragma unroll
    for (int qq = 0; qq < 8; ++qq) {
        const int col = jj + qq * 16;
        const float y = (t[qq] - mu) * rs * gamma[col] + beta[col];
        out_f32[base + row * 128 + col] = y;
        out_f16[base + row * 128 + col] = (f16_t)y;
    }
}

// Kernel 4: q_seed = S @ Wq[2] + bq[2]   (single [1,128]@[128,128] row)
__global__ __launch_bounds__(128) void pma_q_kernel(
    const float* __restrict__ S, const float* __restrict__ Wq,
    const float* __restrict__ bq, float* __restrict__ qs_out)
{
    const int t = threadIdx.x;
    float s = bq[t];
    for (int d = 0; d < 128; ++d) s += S[d] * Wq[d * 128 + t];
    qs_out[t] = s;
}

// Kernel 5: PMA attention (nq=1, mask_a=1): feat_pre = LN(q_s + softmax(qK^T)V)
__global__ __launch_bounds__(128) void pma_attn_kernel(
    const float* __restrict__ qsg, const float* __restrict__ k, const float* __restrict__ v,
    const int* __restrict__ cate, const float* __restrict__ gamma,
    const float* __restrict__ beta, float* __restrict__ out_f32, f16_t* __restrict__ out_f16)
{
    __shared__ float qs[128], ks[16 * 128], vs[16 * 128], dots[64], cm[16], red[8];
    const int b = blockIdx.x, t = threadIdx.x;
    const size_t base = (size_t)b * 2048;
    qs[t] = qsg[t];
#pragma unroll
    for (int rr = 0; rr < 16; ++rr) {
        ks[rr * 128 + t] = k[base + rr * 128 + t];
        vs[rr * 128 + t] = v[base + rr * 128 + t];
    }
    if (t < 16) cm[t] = (cate[b * 16 + t] != -1) ? 1.f : 0.f;
    __syncthreads();
    if (t < 64) {
        const int hh = t >> 4, j = t & 15;
        float s = 0.f;
        for (int d = 0; d < 32; ++d) s += qs[hh * 32 + d] * ks[j * 128 + hh * 32 + d];
        s *= ATT_SCALE;
        if (cm[j] != 1.f) s = NEG_MAX;
        dots[t] = s;
    }
    __syncthreads();
    if (t < 4) {
        float* rp = dots + t * 16;
        float mx = NEG_MAX;
        for (int j = 0; j < 16; ++j) mx = fmaxf(mx, rp[j]);
        float tmp[16], sum = 0.f;
        for (int j = 0; j < 16; ++j) { tmp[j] = __expf(rp[j] - mx); sum += tmp[j]; }
        const float inv = 1.f / sum;
        for (int j = 0; j < 16; ++j) rp[j] = tmp[j] * inv;
    }
    __syncthreads();
    const int hh = t >> 5;
    float o = 0.f;
    for (int j = 0; j < 16; ++j) o += dots[hh * 16 + j] * vs[j * 128 + t];
    const float h1 = qs[t] + o;
    float s1 = h1, s2 = h1 * h1;
#pragma unroll
    for (int m = 16; m >= 1; m >>= 1) { s1 += __shfl_xor(s1, m, 32); s2 += __shfl_xor(s2, m, 32); }
    if ((t & 31) == 0) { red[t >> 5] = s1; red[4 + (t >> 5)] = s2; }
    __syncthreads();
    const float S1 = red[0] + red[1] + red[2] + red[3];
    const float S2 = red[4] + red[5] + red[6] + red[7];
    const float mu = S1 * (1.f / 128.f);
    const float rs = rsqrtf(S2 * (1.f / 128.f) - mu * mu + 1e-5f);
    const float y = (h1 - mu) * rs * gamma[t] + beta[t];
    out_f32[(size_t)b * 128 + t] = y;
    out_f16[(size_t)b * 128 + t] = (f16_t)y;
}

// Kernel 6: cosine scoring against 16 l2-normalized prototypes.
__device__ __forceinline__ float blk_reduce_128(float v, float* red) {
#pragma unroll
    for (int m = 16; m >= 1; m >>= 1) v += __shfl_xor(v, m, 32);
    __syncthreads();
    if ((threadIdx.x & 31) == 0) red[threadIdx.x >> 5] = v;
    __syncthreads();
    return red[0] + red[1] + red[2] + red[3];
}

__global__ __launch_bounds__(128) void score_kernel(
    const float* __restrict__ hx, const float* __restrict__ proto,
    const int* __restrict__ uidx, float* __restrict__ out)
{
    __shared__ float red[4];
    const int b = blockIdx.x, t = threadIdx.x;
    const float hv = hx[(size_t)b * 128 + t];
    const float n2 = blk_reduce_128(hv * hv, red);
    const float invh = 1.f / fmaxf(sqrtf(n2), 1e-12f);
    const float* pb = proto + (size_t)uidx[b] * 16 * 128;
    float total = 0.f;
    for (int p = 0; p < 16; ++p) {
        const float w = pb[p * 128 + t];
        const float dv = blk_reduce_128(hv * w, red);
        const float pn = blk_reduce_128(w * w, red);
        total += dv * invh / fmaxf(sqrtf(pn), 1e-12f);
    }
    if (t == 0) out[b] = total;
}

// ---------------------------------------------------------------------------
extern "C" void kernel_launch(void* const* d_in, const int* in_sizes, int n_in,
                              void* d_out, int out_size, void* d_ws, size_t ws_size,
                              hipStream_t stream) {
    const float* x        = (const float*)d_in[0];
    const int*   cate     = (const int*)d_in[1];
    const int*   uidx     = (const int*)d_in[2];
    const float* w_embed  = (const float*)d_in[3];
    const float* b_embed  = (const float*)d_in[4];
    const float* g_embed  = (const float*)d_in[5];
    const float* be_embed = (const float*)d_in[6];
    const float* Wq = (const float*)d_in[7];   const float* bq = (const float*)d_in[8];
    const float* Wk = (const float*)d_in[9];   const float* bk = (const float*)d_in[10];
    const float* Wv = (const float*)d_in[11];  const float* bv = (const float*)d_in[12];
    const float* Wo = (const float*)d_in[13];  const float* bo = (const float*)d_in[14];
    const float* g0 = (const float*)d_in[15];  const float* be0 = (const float*)d_in[16];
    const float* g1 = (const float*)d_in[17];  const float* be1 = (const float*)d_in[18];
    const float* S  = (const float*)d_in[19];
    const float* proto = (const float*)d_in[20];
    const float* Wt = (const float*)d_in[21];  const float* bt = (const float*)d_in[22];
    float* out = (float*)d_out;

    char* ws = (char*)d_ws;
    float* h_f32    = (float*)(ws + 0);            // 8192*128*4 = 4 MiB
    f16_t* h_f16    = (f16_t*)(ws + 4194304);      // 2 MiB
    float* qb       = (float*)(ws + 6291456);      // 4 MiB
    float* kb       = (float*)(ws + 10485760);     // 4 MiB
    float* vb       = (float*)(ws + 14680064);     // 4 MiB
    float* qsb      = (float*)(ws + 18874368);     // 512 B (pad to 1 KiB)
    float* feat_f32 = (float*)(ws + 18875392);     // 256 KiB
    f16_t* feat_f16 = (f16_t*)(ws + 19137536);     // 128 KiB
    float* hx       = (float*)(ws + 19268608);     // 256 KiB

    embed_gemm_kernel<<<512, 256, 0, stream>>>(x, w_embed, b_embed, g_embed, be_embed,
                                               h_f32, h_f16);
    for (int i = 0; i < 2; ++i) {
        const int wofs = i * 16384, bofs = i * 128;
        gemm128_kernel<0><<<512, 256, 0, stream>>>(h_f16, Wq + wofs, bq + bofs,
                                                   nullptr, nullptr, nullptr, qb, nullptr);
        gemm128_kernel<0><<<512, 256, 0, stream>>>(h_f16, Wk + wofs, bk + bofs,
                                                   nullptr, nullptr, nullptr, kb, nullptr);
        gemm128_kernel<0><<<512, 256, 0, stream>>>(h_f16, Wv + wofs, bv + bofs,
                                                   nullptr, nullptr, nullptr, vb, nullptr);
        sab_attn_kernel<<<512, 256, 0, stream>>>(qb, kb, vb, cate, g0 + bofs, be0 + bofs,
                                                 h_f32, h_f16);
        gemm128_kernel<2><<<512, 256, 0, stream>>>(h_f16, Wo + wofs, bo + bofs,
                                                   g1 + bofs, be1 + bofs, h_f32,
                                                   h_f32, h_f16);
    }
    // PMA (layer 2 weights)
    pma_q_kernel<<<1, 128, 0, stream>>>(S, Wq + 2 * 16384, bq + 2 * 128, qsb);
    gemm128_kernel<0><<<512, 256, 0, stream>>>(h_f16, Wk + 2 * 16384, bk + 2 * 128,
                                               nullptr, nullptr, nullptr, kb, nullptr);
    gemm128_kernel<0><<<512, 256, 0, stream>>>(h_f16, Wv + 2 * 16384, bv + 2 * 128,
                                               nullptr, nullptr, nullptr, vb, nullptr);
    pma_attn_kernel<<<512, 128, 0, stream>>>(qsb, kb, vb, cate, g0 + 2 * 128, be0 + 2 * 128,
                                             feat_f32, feat_f16);
    gemm128_kernel<2><<<32, 256, 0, stream>>>(feat_f16, Wo + 2 * 16384, bo + 2 * 128,
                                              g1 + 2 * 128, be1 + 2 * 128, feat_f32,
                                              feat_f32, feat_f16);
    gemm128_kernel<0><<<32, 256, 0, stream>>>(feat_f16, Wt, bt,
                                              nullptr, nullptr, nullptr, hx, nullptr);
    score_kernel<<<512, 128, 0, stream>>>(hx, proto, uidx, out);
}
